// RNNetwork_44805098832013
// MI455X (gfx1250) — matrix operand order repaired and checked by reading, not compiled, not run on hardware
//
#include <hip/hip_runtime.h>

typedef __attribute__((ext_vector_type(16))) _Float16 v16h;
typedef __attribute__((ext_vector_type(8)))  float    v8f;

__device__ __forceinline__ float sigmoidf(float x) {
    return 1.0f / (1.0f + __expf(-x));
}

// Address (in halves) of element (m,k) of a 16xK f16 A-matrix stored as
// consecutive WMMA 16x16x32 A-fragments (CDNA5 ISA "16-bit A-Matrix 16x32"):
//   lanes 0-15 : M=0-15, VGPR0-3 hold K=0..7,  VGPR4-7 hold K=16..23
//   lanes 16-31: M=0-15, VGPR0-3 hold K=8..15, VGPR4-7 hold K=24..31
// Storage: [kchunk][lane][e] halves, so each lane's fragment is 32 contiguous bytes.
__device__ __forceinline__ int fragAddrA(int m, int k) {
    int c        = k >> 5;           // 32-wide K chunk
    int kk       = k & 31;
    int rem      = kk & 15;
    int lanehalf = rem >> 3;         // 0 -> lanes 0-15, 1 -> lanes 16-31
    int e        = ((kk >> 4) << 3) + (rem & 7);
    int lane     = (lanehalf << 4) + m;
    return ((c << 5) + lane) * 16 + e;
}

// Repack a (K,N) row-major f32 weight matrix into f16 WMMA B-fragments:
// half index i = ((nc*KC + kc)*32 + lane)*16 + e,
// with n = nc*16 + (lane&15), k = kc*32 + (e>=8 ? 16 : 0) + (lane>=16 ? 8 : 0) + (e&7).
__global__ void pack_weights(const float* __restrict__ W, _Float16* __restrict__ P,
                             int K, int N) {
    int total = K * N;
    int KC = K >> 5;
    for (int i = blockIdx.x * blockDim.x + threadIdx.x; i < total;
         i += gridDim.x * blockDim.x) {
        int e    = i & 15;
        int lane = (i >> 4) & 31;
        int t    = i >> 9;               // nc*KC + kc
        int kc   = t % KC;
        int nc   = t / KC;
        int n = (nc << 4) + (lane & 15);
        int k = (kc << 5) + ((e >> 3) << 4) + ((lane >= 16) ? 8 : 0) + (e & 7);
        P[i] = (_Float16)W[k * N + n];
    }
}

__global__ void zero_loss(float* out) { out[0] = 0.0f; }

// Block = 128 threads (4 waves), covers 32 timesteps (two 16-row M-tiles).
// Each wave owns a slice of the N dimension; every loaded B fragment feeds
// TWO WMMAs (one per M-tile) -> halves L2->WGP B traffic vs 16-row blocks.
__global__ __launch_bounds__(128)
void ecm_rnn_kernel(const float* __restrict__ soc_gt, const float* __restrict__ current,
                    const float* __restrict__ vmeas,
                    const float* __restrict__ Zw1, const float* __restrict__ Zb1,
                    const float* __restrict__ Zb2, const float* __restrict__ Zw3,
                    const float* __restrict__ Zb3,
                    const float* __restrict__ Vw1, const float* __restrict__ Vb1,
                    const float* __restrict__ Vb2, const float* __restrict__ Vw3,
                    const float* __restrict__ Vb3,
                    const _Float16* __restrict__ BzP, const _Float16* __restrict__ BvP,
                    float* __restrict__ out_loss, float* __restrict__ out_V,
                    float* __restrict__ out_soc, int T)
{
    // A fragments (activations): 2 M-tiles -> 64KB + 32KB = 96KB LDS.
    __shared__ __attribute__((aligned(32))) _Float16 aZ[2 * 16 * 1024]; // z-net h1
    __shared__ __attribute__((aligned(32))) _Float16 aV[2 * 16 * 512];  // voc h1
    __shared__ float socv[32], siv[32], zAcc[32], vAcc[32], lossP[32];
    __shared__ float Ip_s[32], I_s[32], vm_s[32];

    const int tid  = threadIdx.x;
    const int lane = tid & 31;
    const int wave = tid >> 5;
    const int t0   = blockIdx.x * 32;

    if (tid < 32) {
        int t = t0 + tid;
        float Ip = current[(t == 0) ? 0 : (t - 1)];
        socv[tid] = (t == 0) ? 1.0f : soc_gt[t - 1];
        siv[tid]  = (Ip + 2.0f) * 0.25f;      // (I - CUR_MIN)/(CUR_MAX - CUR_MIN)
        Ip_s[tid] = Ip;
        I_s[tid]  = current[t];
        vm_s[tid] = vmeas[t];
        zAcc[tid] = 0.0f; vAcc[tid] = 0.0f;
    }
    __syncthreads();

    for (int phase = 0; phase < 2; ++phase) {
        // ---- layer 1 (VALU) -> A fragments in LDS ----
        for (int i = tid; i < 32 * 1024; i += 128) {
            int m = i >> 10, k = i & 1023;
            float x = socv[m] * Zw1[k] + siv[m] * Zw1[1024 + k] + Zb1[k];
            aZ[((m >> 4) << 14) + fragAddrA(m & 15, k)] = (_Float16)sigmoidf(x);
        }
        for (int i = tid; i < 32 * 512; i += 128) {
            int m = i >> 9, k = i & 511;
            float x = socv[m] * Vw1[k] + Vb1[k];
            aV[((m >> 4) << 13) + fragAddrA(m & 15, k)] = (_Float16)sigmoidf(x);
        }
        __syncthreads();

        const v16h* apZ = (const v16h*)aZ;   // tile 1 at v16h offset 1024
        const v16h* apV = (const v16h*)aV;   // tile 1 at v16h offset 512
        const int nlo = lane & 15;

        // ---- z-net layer 2: (32x1024)@(1024x512), fused sigmoid + w3 dot ----
        {
            float pz0[8] = {0.f,0.f,0.f,0.f,0.f,0.f,0.f,0.f};
            float pz1[8] = {0.f,0.f,0.f,0.f,0.f,0.f,0.f,0.f};
            for (int nc = wave * 8; nc < wave * 8 + 8; ++nc) {
                int n = (nc << 4) + nlo;
                float bz = Zb2[n];
                v8f c0 = {bz, bz, bz, bz, bz, bz, bz, bz};
                v8f c1 = c0;
                const v16h* bp = ((const v16h*)BzP) + (size_t)nc * 32 * 32;
                #pragma unroll 4
                for (int kc = 0; kc < 32; ++kc) {
                    v16h b  = bp[(kc << 5) + lane];
                    v16h a0 = apZ[(kc << 5) + lane];
                    v16h a1 = apZ[1024 + (kc << 5) + lane];
                    c0 = __builtin_amdgcn_wmma_f32_16x16x32_f16(
                            false, a0, false, b, (short)0, c0, false, false);
                    c1 = __builtin_amdgcn_wmma_f32_16x16x32_f16(
                            false, a1, false, b, (short)0, c1, false, false);
                }
                float w3 = Zw3[n];
                #pragma unroll
                for (int r = 0; r < 8; ++r) {
                    pz0[r] += sigmoidf(c0[r]) * w3;
                    pz1[r] += sigmoidf(c1[r]) * w3;
                }
            }
            #pragma unroll
            for (int r = 0; r < 8; ++r) {
                float v0 = pz0[r], v1 = pz1[r];
                v0 += __shfl_xor(v0, 1, 32);  v1 += __shfl_xor(v1, 1, 32);
                v0 += __shfl_xor(v0, 2, 32);  v1 += __shfl_xor(v1, 2, 32);
                v0 += __shfl_xor(v0, 4, 32);  v1 += __shfl_xor(v1, 4, 32);
                v0 += __shfl_xor(v0, 8, 32);  v1 += __shfl_xor(v1, 8, 32);
                if ((lane & 15) == 0) {
                    int m = r + ((lane >> 4) << 3);
                    atomicAdd(&zAcc[m], v0);
                    atomicAdd(&zAcc[16 + m], v1);
                }
            }
        }

        // ---- voc-net layer 2: (32x512)@(512x256), fused sigmoid + w3 dot ----
        {
            float pv0[8] = {0.f,0.f,0.f,0.f,0.f,0.f,0.f,0.f};
            float pv1[8] = {0.f,0.f,0.f,0.f,0.f,0.f,0.f,0.f};
            for (int nc = wave * 4; nc < wave * 4 + 4; ++nc) {
                int n = (nc << 4) + nlo;
                float bv = Vb2[n];
                v8f c0 = {bv, bv, bv, bv, bv, bv, bv, bv};
                v8f c1 = c0;
                const v16h* bp = ((const v16h*)BvP) + (size_t)nc * 16 * 32;
                #pragma unroll 4
                for (int kc = 0; kc < 16; ++kc) {
                    v16h b  = bp[(kc << 5) + lane];
                    v16h a0 = apV[(kc << 5) + lane];
                    v16h a1 = apV[512 + (kc << 5) + lane];
                    c0 = __builtin_amdgcn_wmma_f32_16x16x32_f16(
                            false, a0, false, b, (short)0, c0, false, false);
                    c1 = __builtin_amdgcn_wmma_f32_16x16x32_f16(
                            false, a1, false, b, (short)0, c1, false, false);
                }
                float w3 = Vw3[n];
                #pragma unroll
                for (int r = 0; r < 8; ++r) {
                    pv0[r] += sigmoidf(c0[r]) * w3;
                    pv1[r] += sigmoidf(c1[r]) * w3;
                }
            }
            #pragma unroll
            for (int r = 0; r < 8; ++r) {
                float v0 = pv0[r], v1 = pv1[r];
                v0 += __shfl_xor(v0, 1, 32);  v1 += __shfl_xor(v1, 1, 32);
                v0 += __shfl_xor(v0, 2, 32);  v1 += __shfl_xor(v1, 2, 32);
                v0 += __shfl_xor(v0, 4, 32);  v1 += __shfl_xor(v1, 4, 32);
                v0 += __shfl_xor(v0, 8, 32);  v1 += __shfl_xor(v1, 8, 32);
                if ((lane & 15) == 0) {
                    int m = r + ((lane >> 4) << 3);
                    atomicAdd(&vAcc[m], v0);
                    atomicAdd(&vAcc[16 + m], v1);
                }
            }
        }
        __syncthreads();

        // ---- scalar ECM physics ----
        if (tid < 32) {
            float Z   = zAcc[tid] + Zb3[0];
            float Voc = vAcc[tid] + Vb3[0];
            if (phase == 0) {
                float Ip = Ip_s[tid];
                float V1 = Voc - Ip * Z;
                float s2 = socv[tid] - Ip * V1 * (1.0f / 20000.0f);
                s2 = (s2 > 1.0f) ? 1.0f : s2;
                s2 = (s2 < 0.0f) ? 1e-10f : s2;
                out_soc[t0 + tid] = s2;
                socv[tid] = s2;
                siv[tid]  = (I_s[tid] + 2.0f) * 0.25f;
                zAcc[tid] = 0.0f; vAcc[tid] = 0.0f;
            } else {
                float I = I_s[tid];
                float V = Voc - I * Z;
                out_V[t0 + tid] = V;
                float d = V - vm_s[tid];
                lossP[tid] = d * d;
            }
        }
        __syncthreads();
    }

    if (tid == 0) {
        float s = 0.0f;
        #pragma unroll
        for (int m = 0; m < 32; ++m) s += lossP[m];
        atomicAdd(out_loss, s * (1.0f / (float)T));
    }
}

extern "C" void kernel_launch(void* const* d_in, const int* in_sizes, int n_in,
                              void* d_out, int out_size, void* d_ws, size_t ws_size,
                              hipStream_t stream) {
    const float* soc_gt  = (const float*)d_in[0];
    const float* current = (const float*)d_in[1];
    const float* vmeas   = (const float*)d_in[2];
    const float* Zw1 = (const float*)d_in[3];
    const float* Zb1 = (const float*)d_in[4];
    const float* Zw2 = (const float*)d_in[5];
    const float* Zb2 = (const float*)d_in[6];
    const float* Zw3 = (const float*)d_in[7];
    const float* Zb3 = (const float*)d_in[8];
    const float* Vw1 = (const float*)d_in[9];
    const float* Vb1 = (const float*)d_in[10];
    const float* Vw2 = (const float*)d_in[11];
    const float* Vb2 = (const float*)d_in[12];
    const float* Vw3 = (const float*)d_in[13];
    const float* Vb3 = (const float*)d_in[14];
    float* out = (float*)d_out;
    const int T = in_sizes[0];            // 65536

    // ws: f16 packed B matrices (1MB + 256KB)
    _Float16* BzP = (_Float16*)d_ws;
    _Float16* BvP = BzP + 1024 * 512;

    pack_weights<<<512, 256, 0, stream>>>(Zw2, BzP, 1024, 512);
    pack_weights<<<128, 256, 0, stream>>>(Vw2, BvP, 512, 256);
    zero_loss<<<1, 1, 0, stream>>>(out);

    // d_out layout: [0]=loss, [1..T]=voltage, [1+T..2T]=soc_hist
    ecm_rnn_kernel<<<T / 32, 128, 0, stream>>>(
        soc_gt, current, vmeas,
        Zw1, Zb1, Zb2, Zw3, Zb3,
        Vw1, Vb1, Vb2, Vw3, Vb3,
        BzP, BvP,
        out, out + 1, out + 1 + T, T);
}
